// LSTMAttentionModel_73194832658577
// MI455X (gfx1250) — compile-verified, matrix operand order checked by reading, not run on hardware
//
#include <hip/hip_runtime.h>
#include <hip/hip_bf16.h>
#include <math.h>

#define BATCH 64
#define SEQ   512
#define FEAT  64
#define BS    (BATCH * SEQ)

typedef float    v8f   __attribute__((ext_vector_type(8)));
typedef _Float16 v16h  __attribute__((ext_vector_type(16)));
typedef _Float16 half8 __attribute__((ext_vector_type(8)));

static __device__ __forceinline__ v16h combine8(half8 lo, half8 hi) {
    return __builtin_shufflevector(lo, hi, 0,1,2,3,4,5,6,7,8,9,10,11,12,13,14,15);
}

// D = A(16x32 f16) * B(32x16 f16) + C (f32 accum), CDNA5 wave32 WMMA.
static __device__ __forceinline__ v8f wmma_f32_16x16x32(v16h a, v16h b, v8f c) {
    return __builtin_amdgcn_wmma_f32_16x16x32_f16(false, a, false, b, (short)0, c, false, false);
}

static __device__ __forceinline__ float sigmoidf_(float x) {
    return 1.0f / (1.0f + __expf(-x));
}

// ---------------------------------------------------------------------------
// Input-projection GEMM: C[BS,N] = A[BS,K](f32) @ W[K,N](f32->f16) + bias[N]
// 4 waves / block, each wave owns one 16x16 output tile (N tiled by blockIdx.y).
// A fragments: per-lane direct global loads + cvt; B fragments from LDS W^T.
// A layout (16-bit 16x32): lane<16 -> row=lane, K={0..7,16..23}; lane>=16 ->
// row=lane-16, K={8..15,24..31}. B assumed symmetric (col per lane).
// ---------------------------------------------------------------------------
template<int K, int N>
__global__ void gemm_in_kernel(const float* __restrict__ A,
                               const float* __restrict__ W,
                               const float* __restrict__ bias,
                               float* __restrict__ C) {
    constexpr int KT = K / 32;
    __shared__ _Float16 wT[N][K];   // transposed, f16
    for (int i = threadIdx.x; i < K * N; i += blockDim.x) {
        int k = i / N, n = i % N;
        wT[n][k] = (_Float16)W[k * N + n];
    }
    __syncthreads();

    const int wid  = threadIdx.x >> 5;
    const int lane = threadIdx.x & 31;
    const int l15  = lane & 15;
    const int hk   = (lane < 16) ? 0 : 8;
    const int row0 = blockIdx.x * 16;
    const int nTile = blockIdx.y * 4 + wid;
    const int col0 = nTile * 16;

    v8f acc = {};
#pragma unroll
    for (int kt = 0; kt < KT; ++kt) {
        const int kb = kt * 32;
        const float* arow = A + (size_t)(row0 + l15) * K + kb + hk;
        half8 alo, ahi;
#pragma unroll
        for (int j = 0; j < 8; ++j) {
            alo[j] = (_Float16)arow[j];
            ahi[j] = (_Float16)arow[16 + j];
        }
        const _Float16* wrow = &wT[col0 + l15][kb + hk];
        half8 blo = *(const half8*)wrow;
        half8 bhi = *(const half8*)(wrow + 16);
        acc = wmma_f32_16x16x32(combine8(alo, ahi), combine8(blo, bhi), acc);
    }
    // D layout: vgpr r, lane<16 -> (m=r, n=lane); lane>=16 -> (m=8+r, n=lane-16)
    const int mb = row0 + ((lane < 16) ? 0 : 8);
    const float bval = bias[col0 + l15];
#pragma unroll
    for (int r = 0; r < 8; ++r)
        C[(size_t)(mb + r) * N + col0 + l15] = acc[r] + bval;
}

// ---------------------------------------------------------------------------
// LSTM recurrence. Block = 16 batch samples, H/16 waves. Each wave owns one
// 16-column slice of the hidden state; since i/f/g/o tiles for those columns
// map to identical lane/VGPR positions, the cell update is register-local.
// w_hh fragments live in VGPRs for all 512 steps; h transposes via LDS.
// G[BS,4H] already holds x@w_ih + b (gate order i,f,g,o).
// ---------------------------------------------------------------------------
template<int H>
__global__ void lstm_rec_kernel(const float* __restrict__ G,
                                const float* __restrict__ Whh,  // [H,4H]
                                float* __restrict__ Hout) {     // [BS,H]
    constexpr int Hpad = (H < 32) ? 32 : H;
    constexpr int KT   = Hpad / 32;
    __shared__ _Float16 whhT[4 * H][Hpad];  // [n][k], zero-padded K
    __shared__ _Float16 h_sh[16][Hpad];     // current hidden state, f16

    for (int i = threadIdx.x; i < 4 * H * Hpad; i += blockDim.x) {
        int n = i / Hpad, k = i % Hpad;
        whhT[n][k] = (k < H) ? (_Float16)Whh[k * 4 * H + n] : (_Float16)0.0f;
    }
    for (int i = threadIdx.x; i < 16 * Hpad; i += blockDim.x)
        (&h_sh[0][0])[i] = (_Float16)0.0f;
    __syncthreads();

    const int wid  = threadIdx.x >> 5;
    const int lane = threadIdx.x & 31;
    const int l15  = lane & 15;
    const int hk   = (lane < 16) ? 0 : 8;
    const int col0 = wid * 16;
    const int s0   = blockIdx.x * 16;
    const int mb   = (lane < 16) ? 0 : 8;

    // Resident w_hh B-fragments: [gate][ktile]
    v16h bfrag[4][KT];
#pragma unroll
    for (int g = 0; g < 4; ++g)
#pragma unroll
        for (int kt = 0; kt < KT; ++kt) {
            const _Float16* p = &whhT[g * H + col0 + l15][kt * 32 + hk];
            bfrag[g][kt] = combine8(*(const half8*)p, *(const half8*)(p + 16));
        }

    v8f c = {};
#pragma unroll 1
    for (int t = 0; t < SEQ; ++t) {
        v16h afrag[KT];
#pragma unroll
        for (int kt = 0; kt < KT; ++kt) {
            const _Float16* p = &h_sh[l15][kt * 32 + hk];
            afrag[kt] = combine8(*(const half8*)p, *(const half8*)(p + 16));
        }
        __syncthreads();  // all waves done reading h before it is rewritten

        v8f acc[4] = {v8f{}, v8f{}, v8f{}, v8f{}};
#pragma unroll
        for (int g = 0; g < 4; ++g)
#pragma unroll
            for (int kt = 0; kt < KT; ++kt)
                acc[g] = wmma_f32_16x16x32(afrag[kt], bfrag[g][kt], acc[g]);

#pragma unroll
        for (int r = 0; r < 8; ++r) {
            const size_t row = (size_t)(s0 + mb + r) * SEQ + t;
            const float* gp = G + row * (4 * H) + col0 + l15;
            __builtin_prefetch(gp + 4 * H, 0, 0);  // next timestep's gates
            float iv = acc[0][r] + gp[0 * H];
            float fv = acc[1][r] + gp[1 * H];
            float gv = acc[2][r] + gp[2 * H];
            float ov = acc[3][r] + gp[3 * H];
            float cn = sigmoidf_(fv) * c[r] + sigmoidf_(iv) * tanhf(gv);
            c[r] = cn;
            float hv = sigmoidf_(ov) * tanhf(cn);
            Hout[row * H + col0 + l15] = hv;
            h_sh[mb + r][col0 + l15] = (_Float16)hv;
        }
        __syncthreads();  // h writes visible before next step's A loads
    }
}

// ---------------------------------------------------------------------------
__global__ void posenc_add_kernel(float* __restrict__ h) {  // [BS,16] in place
    int idx = blockIdx.x * blockDim.x + threadIdx.x;
    if (idx >= BS * 16) return;
    int d = idx & 15;
    int t = (idx >> 4) % SEQ;
    float div = __expf(-(float)(d & ~1) * (9.210340372f / 16.0f));
    float ang = (float)t * div;
    h[idx] += (d & 1) ? __cosf(ang) : __sinf(ang);
}

__global__ void qkv_kernel(const float* __restrict__ h,
                           const float* __restrict__ wq, const float* __restrict__ bq,
                           const float* __restrict__ wk, const float* __restrict__ bk,
                           const float* __restrict__ wv, const float* __restrict__ bv,
                           float* __restrict__ Q, float* __restrict__ K,
                           float* __restrict__ V) {
    int tok = blockIdx.x * blockDim.x + threadIdx.x;
    if (tok >= BS) return;
    float x[16];
#pragma unroll
    for (int i = 0; i < 16; ++i) x[i] = h[tok * 16 + i];
#pragma unroll
    for (int j = 0; j < 16; ++j) {
        float q = bq[j], k = bk[j], v = bv[j];
#pragma unroll
        for (int i = 0; i < 16; ++i) {
            float xv = x[i];
            q += xv * wq[i * 16 + j];
            k += xv * wk[i * 16 + j];
            v += xv * wv[i * 16 + j];
        }
        Q[tok * 16 + j] = q; K[tok * 16 + j] = k; V[tok * 16 + j] = v;
    }
}

// One block per (batch, head); K/V staged in LDS; online softmax. hd = 2.
__global__ void attn_kernel(const float* __restrict__ Q, const float* __restrict__ K,
                            const float* __restrict__ V, float* __restrict__ O) {
    const int b = blockIdx.x >> 3;
    const int hd = blockIdx.x & 7;
    __shared__ float ks[SEQ][2];
    __shared__ float vs[SEQ][2];
    const int t = threadIdx.x;
    const size_t base = (size_t)b * SEQ * 16 + hd * 2;
    ks[t][0] = K[base + t * 16];     ks[t][1] = K[base + t * 16 + 1];
    vs[t][0] = V[base + t * 16];     vs[t][1] = V[base + t * 16 + 1];
    __syncthreads();
    const float q0 = Q[base + t * 16], q1 = Q[base + t * 16 + 1];
    const float scale = 0.7071067811865475f;  // 1/sqrt(2)
    float m = -1e30f, l = 0.0f, a0 = 0.0f, a1 = 0.0f;
#pragma unroll 4
    for (int j = 0; j < SEQ; ++j) {
        float s  = (q0 * ks[j][0] + q1 * ks[j][1]) * scale;
        float nm = fmaxf(m, s);
        float cr = __expf(m - nm);
        float p  = __expf(s - nm);
        l  = l * cr + p;
        a0 = a0 * cr + p * vs[j][0];
        a1 = a1 * cr + p * vs[j][1];
        m  = nm;
    }
    O[base + t * 16]     = a0 / l;
    O[base + t * 16 + 1] = a1 / l;
}

static __device__ __forceinline__ void ln16(const float* r, const float* g,
                                            const float* be, float* out) {
    float m = 0.0f;
#pragma unroll
    for (int i = 0; i < 16; ++i) m += r[i];
    m *= (1.0f / 16.0f);
    float v = 0.0f;
#pragma unroll
    for (int i = 0; i < 16; ++i) { float d = r[i] - m; v += d * d; }
    v *= (1.0f / 16.0f);
    float inv = rsqrtf(v + 1e-5f);
#pragma unroll
    for (int i = 0; i < 16; ++i) out[i] = (r[i] - m) * inv * g[i] + be[i];
}

__global__ void proj_ln_kernel(const float* __restrict__ A, const float* __restrict__ hin,
                               const float* __restrict__ wo, const float* __restrict__ bo,
                               const float* __restrict__ g,  const float* __restrict__ be,
                               float* __restrict__ out) {
    int tok = blockIdx.x * blockDim.x + threadIdx.x;
    if (tok >= BS) return;
    float a[16], r[16];
#pragma unroll
    for (int i = 0; i < 16; ++i) a[i] = A[tok * 16 + i];
#pragma unroll
    for (int j = 0; j < 16; ++j) {
        float o = bo[j];
#pragma unroll
        for (int i = 0; i < 16; ++i) o += a[i] * wo[i * 16 + j];
        r[j] = o + hin[tok * 16 + j];
    }
    ln16(r, g, be, out + tok * 16);
}

__global__ void ff_ln_kernel(const float* __restrict__ X,
                             const float* __restrict__ w1, const float* __restrict__ b1,
                             const float* __restrict__ w2, const float* __restrict__ b2,
                             const float* __restrict__ g,  const float* __restrict__ be,
                             float* __restrict__ out) {
    int tok = blockIdx.x * blockDim.x + threadIdx.x;
    if (tok >= BS) return;
    float x[16], u[32], r[16];
#pragma unroll
    for (int i = 0; i < 16; ++i) x[i] = X[tok * 16 + i];
#pragma unroll
    for (int j = 0; j < 32; ++j) {
        float s = b1[j];
#pragma unroll
        for (int i = 0; i < 16; ++i) s += x[i] * w1[i * 32 + j];
        u[j] = fmaxf(s, 0.0f);
    }
#pragma unroll
    for (int j = 0; j < 16; ++j) {
        float s = b2[j];
#pragma unroll
        for (int i = 0; i < 32; ++i) s += u[i] * w2[i * 16 + j];
        r[j] = s + x[j];
    }
    ln16(r, g, be, out + tok * 16);
}

// One block per batch sample: seq-softmax pooling (LDS tree reductions +
// ds_add_f32 atomics) then the tiny classifier head.
__global__ void pool_cls_kernel(const float* __restrict__ X,
                                const float* __restrict__ wp,  const float* __restrict__ bp,
                                const float* __restrict__ wc1, const float* __restrict__ bc1,
                                const float* __restrict__ wc2, const float* __restrict__ bc2,
                                float* __restrict__ out) {
    const int b = blockIdx.x;
    const int t = threadIdx.x;
    __shared__ float red[SEQ];
    __shared__ float pooled[16];
    const float* xr = X + ((size_t)b * SEQ + t) * 16;
    float s = bp[0];
#pragma unroll
    for (int i = 0; i < 16; ++i) s += xr[i] * wp[i];
    red[t] = s;
    __syncthreads();
    for (int off = SEQ / 2; off > 0; off >>= 1) {
        if (t < off) red[t] = fmaxf(red[t], red[t + off]);
        __syncthreads();
    }
    float mx = red[0];
    __syncthreads();
    float e = __expf(s - mx);
    red[t] = e;
    __syncthreads();
    for (int off = SEQ / 2; off > 0; off >>= 1) {
        if (t < off) red[t] += red[t + off];
        __syncthreads();
    }
    float sum = red[0];
    __syncthreads();
    if (t < 16) pooled[t] = 0.0f;
    __syncthreads();
    float w = e / sum;
#pragma unroll
    for (int i = 0; i < 16; ++i) atomicAdd(&pooled[i], w * xr[i]);
    __syncthreads();
    if (t == 0) {
        float hc[8];
#pragma unroll
        for (int j = 0; j < 8; ++j) {
            float a = bc1[j];
#pragma unroll
            for (int i = 0; i < 16; ++i) a += pooled[i] * wc1[i * 8 + j];
            hc[j] = fmaxf(a, 0.0f);
        }
        float lg[3], mm = -1e30f;
#pragma unroll
        for (int j = 0; j < 3; ++j) {
            float a = bc2[j];
#pragma unroll
            for (int i = 0; i < 8; ++i) a += hc[i] * wc2[i * 3 + j];
            lg[j] = a; mm = fmaxf(mm, a);
        }
        float den = 0.0f;
#pragma unroll
        for (int j = 0; j < 3; ++j) { lg[j] = __expf(lg[j] - mm); den += lg[j]; }
#pragma unroll
        for (int j = 0; j < 3; ++j) out[b * 3 + j] = lg[j] / den;
    }
}

// ---------------------------------------------------------------------------
extern "C" void kernel_launch(void* const* d_in, const int* in_sizes, int n_in,
                              void* d_out, int out_size, void* d_ws, size_t ws_size,
                              hipStream_t stream) {
    const float* x     = (const float*)d_in[0];
    const float* w_ih1 = (const float*)d_in[1];
    const float* w_hh1 = (const float*)d_in[2];
    const float* b1    = (const float*)d_in[3];
    const float* w_ih2 = (const float*)d_in[4];
    const float* w_hh2 = (const float*)d_in[5];
    const float* b2    = (const float*)d_in[6];
    const float* w_ih3 = (const float*)d_in[7];
    const float* w_hh3 = (const float*)d_in[8];
    const float* b3    = (const float*)d_in[9];
    const float* wq = (const float*)d_in[10]; const float* bq = (const float*)d_in[11];
    const float* wk = (const float*)d_in[12]; const float* bk = (const float*)d_in[13];
    const float* wv = (const float*)d_in[14]; const float* bv = (const float*)d_in[15];
    const float* wo = (const float*)d_in[16]; const float* bo = (const float*)d_in[17];
    const float* w_ff1 = (const float*)d_in[18]; const float* b_ff1 = (const float*)d_in[19];
    const float* w_ff2 = (const float*)d_in[20]; const float* b_ff2 = (const float*)d_in[21];
    const float* g1  = (const float*)d_in[22]; const float* be1 = (const float*)d_in[23];
    const float* g2  = (const float*)d_in[24]; const float* be2 = (const float*)d_in[25];
    const float* wp  = (const float*)d_in[26]; const float* bp  = (const float*)d_in[27];
    const float* wc1 = (const float*)d_in[28]; const float* bc1 = (const float*)d_in[29];
    const float* wc2 = (const float*)d_in[30]; const float* bc2 = (const float*)d_in[31];
    float* out = (float*)d_out;

    float* ws = (float*)d_ws;
    float* G   = ws;                 // BS*256 (reused by all 3 layers)
    float* H1  = G  + (size_t)BS * 256;   // BS*64
    float* H2  = H1 + (size_t)BS * 64;    // BS*32
    float* h   = H2 + (size_t)BS * 32;    // BS*16
    float* Qb  = h  + (size_t)BS * 16;
    float* Kb  = Qb + (size_t)BS * 16;
    float* Vb  = Kb + (size_t)BS * 16;
    float* Ab  = Vb + (size_t)BS * 16;
    float* T1  = Ab + (size_t)BS * 16;
    float* T2  = T1 + (size_t)BS * 16;

    // Layer 1: G = x @ w_ih1 + b1 ; recurrence
    gemm_in_kernel<64, 256><<<dim3(BS / 16, 4), 128, 0, stream>>>(x, w_ih1, b1, G);
    lstm_rec_kernel<64><<<BATCH / 16, 128, 0, stream>>>(G, w_hh1, H1);
    // Layer 2
    gemm_in_kernel<64, 128><<<dim3(BS / 16, 2), 128, 0, stream>>>(H1, w_ih2, b2, G);
    lstm_rec_kernel<32><<<BATCH / 16, 64, 0, stream>>>(G, w_hh2, H2);
    // Layer 3
    gemm_in_kernel<32, 64><<<dim3(BS / 16, 1), 128, 0, stream>>>(H2, w_ih3, b3, G);
    lstm_rec_kernel<16><<<BATCH / 16, 32, 0, stream>>>(G, w_hh3, h);

    posenc_add_kernel<<<(BS * 16 + 255) / 256, 256, 0, stream>>>(h);
    qkv_kernel<<<(BS + 255) / 256, 256, 0, stream>>>(h, wq, bq, wk, bk, wv, bv, Qb, Kb, Vb);
    attn_kernel<<<BATCH * 8, SEQ, 0, stream>>>(Qb, Kb, Vb, Ab);
    proj_ln_kernel<<<(BS + 255) / 256, 256, 0, stream>>>(Ab, h, wo, bo, g1, be1, T1);
    ff_ln_kernel<<<(BS + 255) / 256, 256, 0, stream>>>(T1, w_ff1, b_ff1, w_ff2, b_ff2, g2, be2, T2);
    pool_cls_kernel<<<BATCH, SEQ, 0, stream>>>(T2, wp, bp, wc1, bc1, wc2, bc2, out);
    (void)in_sizes; (void)n_in; (void)out_size; (void)ws_size;
}
